// CausalSelfAttention_88673894793831
// MI455X (gfx1250) — matrix operand order repaired and checked by reading, not compiled, not run on hardware
//
#include <hip/hip_runtime.h>
#include <hip/hip_bf16.h>
#include <math.h>

typedef __attribute__((ext_vector_type(16))) _Float16 v16h;
typedef __attribute__((ext_vector_type(8)))  float    v8f;
typedef __attribute__((ext_vector_type(4)))  float    fvec4;

#define B_   4
#define T_   827
#define C_   1024
#define NH_  16
#define HS_  64
#define M_   (B_*T_)        /* 3308 */
#define SCALE_ 0.125f       /* 1/sqrt(64) */
#define NEGBIG -1.0e30f

static __device__ __forceinline__ int imin(int a, int b) { return a < b ? a : b; }

static __device__ __forceinline__ float blkMax(float x, float* red) {
  int t = threadIdx.x;
  red[t] = x; __syncthreads();
  #pragma unroll
  for (int s = 128; s > 0; s >>= 1) { if (t < s) red[t] = fmaxf(red[t], red[t + s]); __syncthreads(); }
  float r = red[0]; __syncthreads(); return r;
}
static __device__ __forceinline__ float blkSum(float x, float* red) {
  int t = threadIdx.x;
  red[t] = x; __syncthreads();
  #pragma unroll
  for (int s = 128; s > 0; s >>= 1) { if (t < s) red[t] += red[t + s]; __syncthreads(); }
  float r = red[0]; __syncthreads(); return r;
}

// ---------------------------------------------------------------------------
// Y[M,N] = X[M,K] @ W[N,K]^T + bias.  f32 in/out, f16 WMMA compute.
// 256 threads (8 waves). Block tile 128(M) x 64(N), K step 32.
// Wave w owns rows [w*16, w*16+16) x all 4 N-subtiles -> 4 accumulators.
// ---------------------------------------------------------------------------
__global__ __launch_bounds__(256) void gemm_bias_wmma(
    const float* __restrict__ X, const float* __restrict__ W,
    const float* __restrict__ bias, float* __restrict__ Y,
    int Mdim, int N, int K)
{
  __shared__ _Float16 As[128 * 32];   // 8 KB
  __shared__ _Float16 Bs[64 * 32];    // 4 KB
  const int t = threadIdx.x;
  const int wave = t >> 5, lane = t & 31;
  const int m0 = blockIdx.y * 128;
  const int n0 = blockIdx.x * 64;

  v8f acc[4] = {};

  const int arow  = t >> 1, acol0 = (t & 1) * 16;  // A tile: 128 x 32, 16 elems/thread
  const int brow  = t >> 2, bcol0 = (t & 3) * 8;   // B tile: 64 x 32,  8 elems/thread

  for (int k0 = 0; k0 < K; k0 += 32) {
    if (k0 + 32 < K) {  // gfx1250 global_prefetch_b8 on the next K slice
      __builtin_prefetch(&X[(size_t)imin(m0 + arow, Mdim - 1) * K + k0 + 32 + acol0], 0, 1);
      __builtin_prefetch(&W[(size_t)(n0 + brow) * K + k0 + 32 + bcol0], 0, 1);
    }
    {   // stage A (f32 -> f16)
      const float* src = &X[(size_t)imin(m0 + arow, Mdim - 1) * K + k0 + acol0];
      _Float16* dst = &As[arow * 32 + acol0];
      #pragma unroll
      for (int i = 0; i < 4; i++) {
        fvec4 x4 = *(const fvec4*)(src + i * 4);
        dst[i * 4 + 0] = (_Float16)x4.x; dst[i * 4 + 1] = (_Float16)x4.y;
        dst[i * 4 + 2] = (_Float16)x4.z; dst[i * 4 + 3] = (_Float16)x4.w;
      }
    }
    {   // stage B (f32 -> f16)
      const float* src = &W[(size_t)(n0 + brow) * K + k0 + bcol0];
      _Float16* dst = &Bs[brow * 32 + bcol0];
      #pragma unroll
      for (int i = 0; i < 2; i++) {
        fvec4 x4 = *(const fvec4*)(src + i * 4);
        dst[i * 4 + 0] = (_Float16)x4.x; dst[i * 4 + 1] = (_Float16)x4.y;
        dst[i * 4 + 2] = (_Float16)x4.z; dst[i * 4 + 3] = (_Float16)x4.w;
      }
    }
    __syncthreads();

    v16h a = *(const v16h*)&As[(wave * 16 + (lane & 15)) * 32 + (lane >> 4) * 16];
    #pragma unroll
    for (int nt = 0; nt < 4; nt++) {
      v16h b = *(const v16h*)&Bs[(nt * 16 + (lane & 15)) * 32 + (lane >> 4) * 16];
      acc[nt] = __builtin_amdgcn_wmma_f32_16x16x32_f16(
          false, a, false, b, (short)0, acc[nt], false, false);
    }
    __syncthreads();
  }

  // epilogue: C/D layout -> VGPR r holds M=r (lanes 0-15) / M=r+8 (lanes 16-31)
  #pragma unroll
  for (int nt = 0; nt < 4; nt++) {
    const int col = n0 + nt * 16 + (lane & 15);
    const float bv = bias[col];
    #pragma unroll
    for (int r = 0; r < 8; r++) {
      int row = m0 + wave * 16 + r + ((lane >= 16) ? 8 : 0);
      if (row < Mdim) Y[(size_t)row * N + col] = acc[nt][r] + bv;
    }
  }
}

// ---------------------------------------------------------------------------
// (B,T,C) f32  ->  (B,NH,T,HS) f16 head-major layout
// ---------------------------------------------------------------------------
__global__ __launch_bounds__(256) void pack_heads(const float* __restrict__ src,
                                                  _Float16* __restrict__ dst)
{
  size_t i = (size_t)blockIdx.x * 256 + threadIdx.x;
  if (i >= (size_t)M_ * C_) return;
  int c = (int)(i % C_);
  size_t bt = i / C_;
  int tt = (int)(bt % T_);
  int b  = (int)(bt / T_);
  int hh = c / HS_, d = c % HS_;
  dst[(((size_t)(b * NH_ + hh)) * T_ + tt) * HS_ + d] = (_Float16)src[i];
}

// ---------------------------------------------------------------------------
// att[b,h,q,k] = SCALE * <q_row, k_row> + h[b,h,q,k]
// blockIdx: x = q-tile(32), y = k-tile(64), z = b*NH+h.  8 waves, 1 tile each.
// ---------------------------------------------------------------------------
__global__ __launch_bounds__(256) void att_score_wmma(
    const _Float16* __restrict__ qh, const _Float16* __restrict__ kh,
    const float* __restrict__ hbias, float* __restrict__ att)
{
  const int bh = blockIdx.z;
  const int t = threadIdx.x, wave = t >> 5, lane = t & 31;
  const int qbase = blockIdx.x * 32 + (wave >> 2) * 16;
  const int kbase = blockIdx.y * 64 + (wave & 3) * 16;
  const size_t hb = (size_t)bh * T_;

  v8f acc = {};
  #pragma unroll
  for (int s = 0; s < 2; s++) {          // HS = 64 -> two K=32 steps
    int col0 = s * 32 + (lane >> 4) * 16;
    int qr = imin(qbase + (lane & 15), T_ - 1);
    int kr = imin(kbase + (lane & 15), T_ - 1);
    v16h a = *(const v16h*)&qh[(hb + qr) * HS_ + col0];
    v16h b = *(const v16h*)&kh[(hb + kr) * HS_ + col0];
    acc = __builtin_amdgcn_wmma_f32_16x16x32_f16(
        false, a, false, b, (short)0, acc, false, false);
  }
  const int col = kbase + (lane & 15);
  #pragma unroll
  for (int r = 0; r < 8; r++) {
    int row = qbase + r + ((lane >= 16) ? 8 : 0);
    if (row < T_ && col < T_) {
      size_t idx = (hb + row) * T_ + col;
      att[idx] = acc[r] * SCALE_ + hbias[idx];
    }
  }
}

// ---------------------------------------------------------------------------
// in-place softmax over a 256-wide column stripe of one row, with optional
// fused post-scale by f[b,i,j]*bm[b,j,i].  grid: x = 256 rows, y = B*NH
// ---------------------------------------------------------------------------
__global__ __launch_bounds__(256) void softmax_scale_region(
    float* att, const float* __restrict__ f, const float* __restrict__ bm,
    int q0, int c0, int do_scale)
{
  __shared__ float red[256];
  const int t = threadIdx.x, i = blockIdx.x, bh = blockIdx.y;
  size_t base = ((size_t)bh * T_ + q0 + i) * T_ + c0;
  float x = att[base + t];
  float mx = blkMax(x, red);
  float e = __expf(x - mx);
  float s = blkSum(e, red);
  float out = e / s;
  if (do_scale) {
    const int b = bh >> 4;
    out *= f[((size_t)b * 256 + i) * 256 + t] *
           bm[((size_t)b * 256 + t) * 256 + i];
  }
  att[base + t] = out;
}

// ---------------------------------------------------------------------------
// bi_epi_ratio[b,i] = 256 * sum_{h,j}(r3*m) / (sum_j(m) * sum_{h,j}(r3))
// (mask is head-independent so the per-head means collapse algebraically)
// ---------------------------------------------------------------------------
__global__ __launch_bounds__(256) void bi_epi_kernel(
    const float* __restrict__ att, const float* __restrict__ f12,
    const float* __restrict__ b12, float* __restrict__ ratio_out)
{
  __shared__ float red[256];
  const int b = blockIdx.y, i = blockIdx.x, t = threadIdx.x;
  float m = (f12[((size_t)b * 256 + i) * 256 + t] *
             b12[((size_t)b * 256 + t) * 256 + i] >= 0.1f) ? 1.f : 0.f;
  float s_all = 0.f;
  for (int hh = 0; hh < NH_; hh++) {
    size_t base = (((size_t)(b * NH_ + hh)) * T_ + 571 + i) * T_ + 286;
    s_all += att[base + t];
  }
  float Sall = blkSum(s_all, red);
  float Smsk = blkSum(s_all * m, red);
  float Mc   = blkSum(m, red);
  if (t == 0) ratio_out[b * 256 + i] = (256.f * Smsk) / (Mc * Sall);
}

// ---------------------------------------------------------------------------
// att[bh, q0+i, c0+j] *= f[b,i,j] * b[b,j,i]
// ---------------------------------------------------------------------------
__global__ __launch_bounds__(256) void scale_region(
    float* att, const float* __restrict__ f, const float* __restrict__ bm,
    int q0, int c0)
{
  const int bh = blockIdx.y, i = blockIdx.x, t = threadIdx.x, b = bh >> 4;
  size_t idx = ((size_t)bh * T_ + q0 + i) * T_ + c0 + t;
  att[idx] *= f[((size_t)b * 256 + i) * 256 + t] *
              bm[((size_t)b * 256 + t) * 256 + i];
}

// ---------------------------------------------------------------------------
// final causal softmax per row, with the two per-segment softmax overrides
// for rows >= 571.  In place.  grid: x = T rows, y = B*NH.  4 cols/thread.
// ---------------------------------------------------------------------------
__global__ __launch_bounds__(256) void final_row_softmax(float* att)
{
  __shared__ float red[256];
  const int row = blockIdx.x;
  const int t = threadIdx.x;
  size_t base = ((size_t)blockIdx.y * T_ + row) * T_;

  float v[4]; int cols[4];
  #pragma unroll
  for (int u = 0; u < 4; u++) {
    int c = t + u * 256; cols[u] = c;
    v[u] = (c < T_) ? att[base + c] : 0.f;
  }

  const bool special = (row >= 571);   // uniform across the block
  float mA = 0.f, sA = 1.f, mB = 0.f, sB = 1.f;
  if (special) {
    float lx = NEGBIG;
    #pragma unroll
    for (int u = 0; u < 4; u++) if (cols[u] < 256) lx = fmaxf(lx, v[u]);
    mA = blkMax(lx, red);
    float ls = 0.f;
    #pragma unroll
    for (int u = 0; u < 4; u++) if (cols[u] < 256) ls += __expf(v[u] - mA);
    sA = blkSum(ls, red);

    lx = NEGBIG;
    #pragma unroll
    for (int u = 0; u < 4; u++) if (cols[u] >= 286 && cols[u] < 542) lx = fmaxf(lx, v[u]);
    mB = blkMax(lx, red);
    ls = 0.f;
    #pragma unroll
    for (int u = 0; u < 4; u++) if (cols[u] >= 286 && cols[u] < 542) ls += __expf(v[u] - mB);
    sB = blkSum(ls, red);
  }

  float p[4];
  #pragma unroll
  for (int u = 0; u < 4; u++) {
    int c = cols[u];
    float pv;
    if (special && c < 256)                  pv = __expf(v[u] - mA) / sA;
    else if (special && c >= 286 && c < 542) pv = __expf(v[u] - mB) / sB;
    else                                     pv = (c <= row) ? v[u] : NEGBIG;
    p[u] = (c < T_) ? pv : NEGBIG;
  }

  float lx = NEGBIG;
  #pragma unroll
  for (int u = 0; u < 4; u++) lx = fmaxf(lx, p[u]);
  float mx = blkMax(lx, red);
  float ls = 0.f;
  #pragma unroll
  for (int u = 0; u < 4; u++) if (cols[u] < T_) ls += __expf(p[u] - mx);
  float sm = blkSum(ls, red);
  #pragma unroll
  for (int u = 0; u < 4; u++)
    if (cols[u] < T_) att[base + cols[u]] = __expf(p[u] - mx) / sm;
}

// ---------------------------------------------------------------------------
// y32[b, q, h*64+d] = sum_k aw[bh,q,k] * v[bh,k,d]   (head transpose fused in)
// blockIdx: x = q-tile(32), z = b*NH+h.  8 waves: (qsub, d-subtile) grid 2x4.
// aw tile (32x32) is converted f32->f16 once per block into LDS; each wave
// then reads aligned v16h fragments from LDS (ds_load_b128 path).
// ---------------------------------------------------------------------------
__global__ __launch_bounds__(256) void av_wmma(
    const float* __restrict__ att, const _Float16* __restrict__ vh,
    float* __restrict__ y32)
{
  __shared__ _Float16 Aw[32 * 32];   // 2 KB
  const int bh = blockIdx.z;
  const int t = threadIdx.x, wave = t >> 5, lane = t & 31;
  const int q0 = blockIdx.x * 32;
  const int qbase = q0 + (wave >> 2) * 16;
  const int dbase = (wave & 3) * 16;
  const int b = bh >> 4, hh = bh & 15;
  const size_t hb = (size_t)bh * T_;

  // cooperative staging assignment: thread t loads row t/8, cols (t%8)*4 ..+4
  const int lrow  = t >> 3;
  const int lcol0 = (t & 7) * 4;
  const int grow  = imin(q0 + lrow, T_ - 1);
  const float* arow = &att[(hb + grow) * T_];

  v8f acc = {};
  for (int kk = 0; kk < T_; kk += 32) {
    if (kk + 32 < T_)
      __builtin_prefetch(&arow[kk + 32 + lcol0], 0, 1);
    #pragma unroll
    for (int i = 0; i < 4; i++) {
      int c = kk + lcol0 + i;
      Aw[lrow * 32 + lcol0 + i] = (c < T_) ? (_Float16)arow[c] : (_Float16)0.f;
    }
    __syncthreads();

    v16h a = *(const v16h*)&Aw[((wave >> 2) * 16 + (lane & 15)) * 32 + (lane >> 4) * 16];
    v16h bfr;
    #pragma unroll
    for (int s = 0; s < 16; s++) {
      int key = imin(kk + (lane >> 4) * 16 + s, T_ - 1);
      bfr[s] = vh[(hb + key) * HS_ + dbase + (lane & 15)];
    }
    acc = __builtin_amdgcn_wmma_f32_16x16x32_f16(
        false, a, false, bfr, (short)0, acc, false, false);
    __syncthreads();
  }

  const int col = dbase + (lane & 15);
  #pragma unroll
  for (int r = 0; r < 8; r++) {
    int row = qbase + r + ((lane >= 16) ? 8 : 0);
    if (row < T_)
      y32[((size_t)b * T_ + row) * C_ + hh * HS_ + col] = acc[r];
  }
}

// ---------------------------------------------------------------------------
extern "C" void kernel_launch(void* const* d_in, const int* in_sizes, int n_in,
                              void* d_out, int out_size, void* d_ws, size_t ws_size,
                              hipStream_t stream)
{
  (void)in_sizes; (void)n_in; (void)out_size; (void)ws_size;
  const float* x   = (const float*)d_in[0];
  const float* xkv = (const float*)d_in[1];
  const float* h   = (const float*)d_in[2];
  const float* f01 = (const float*)d_in[3];
  const float* f02 = (const float*)d_in[4];
  const float* f12 = (const float*)d_in[5];
  const float* b01 = (const float*)d_in[6];
  const float* b02 = (const float*)d_in[7];
  const float* b12 = (const float*)d_in[8];
  const float* Wq  = (const float*)d_in[9];
  const float* Wk  = (const float*)d_in[10];
  const float* Wv  = (const float*)d_in[11];
  const float* Wp  = (const float*)d_in[12];
  const float* bq  = (const float*)d_in[13];
  const float* bk  = (const float*)d_in[14];
  const float* bv  = (const float*)d_in[15];
  const float* bp  = (const float*)d_in[16];

  const size_t MC = (size_t)M_ * C_;
  float* y_out     = (float*)d_out;
  float* ratio_out = y_out + MC;

  // workspace layout
  float*    q32 = (float*)d_ws;
  float*    k32 = q32 + MC;
  float*    v32 = k32 + MC;
  float*    y32 = v32 + MC;
  _Float16* qh  = (_Float16*)(y32 + MC);
  _Float16* kh  = qh + MC;
  _Float16* vh  = kh + MC;
  float*    att = (float*)(vh + MC);   // B*NH*T*T floats

  dim3 gGemm(C_ / 64, (M_ + 127) / 128);
  gemm_bias_wmma<<<gGemm, 256, 0, stream>>>(x,   Wq, bq, q32, M_, C_, C_);
  gemm_bias_wmma<<<gGemm, 256, 0, stream>>>(xkv, Wk, bk, k32, M_, C_, C_);
  gemm_bias_wmma<<<gGemm, 256, 0, stream>>>(xkv, Wv, bv, v32, M_, C_, C_);

  dim3 gPack((unsigned)((MC + 255) / 256));
  pack_heads<<<gPack, 256, 0, stream>>>(q32, qh);
  pack_heads<<<gPack, 256, 0, stream>>>(k32, kh);
  pack_heads<<<gPack, 256, 0, stream>>>(v32, vh);

  dim3 gAtt((T_ + 31) / 32, (T_ + 63) / 64, B_ * NH_);
  att_score_wmma<<<gAtt, 256, 0, stream>>>(qh, kh, h, att);

  dim3 gSm(256, B_ * NH_);
  // regions 1 & 2: softmax with fused (b*f) scale
  softmax_scale_region<<<gSm, 256, 0, stream>>>(att, f01, b01, 285, 0,   1);
  softmax_scale_region<<<gSm, 256, 0, stream>>>(att, f02, b02, 571, 0,   1);
  // region 3: softmax only (bi-epi ratio needs the pre-scale values)
  softmax_scale_region<<<gSm, 256, 0, stream>>>(att, f12, b12, 571, 286, 0);

  dim3 gRatio(256, B_);
  bi_epi_kernel<<<gRatio, 256, 0, stream>>>(att, f12, b12, ratio_out);

  scale_region<<<gSm, 256, 0, stream>>>(att, f12, b12, 571, 286);

  dim3 gRow(T_, B_ * NH_);
  final_row_softmax<<<gRow, 256, 0, stream>>>(att);

  dim3 gAv((T_ + 31) / 32, 1, B_ * NH_);
  av_wmma<<<gAv, 256, 0, stream>>>(att, vh, y32);

  gemm_bias_wmma<<<gGemm, 256, 0, stream>>>(y32, Wp, bp, y_out, M_, C_, C_);
}
